// GraphTransformer_29300266893495
// MI455X (gfx1250) — matrix operand order repaired and checked by reading, not compiled, not run on hardware
//
#include <hip/hip_runtime.h>
#include <math.h>

// ---------------------------------------------------------------------------
// Problem constants (match reference)
// ---------------------------------------------------------------------------
#define NN      50000
#define EE      800000
#define D_IN    16
#define DM      64
#define HH      8
#define DHH     8
#define LL      2
#define EPSBN   1e-5f
#define INV_SCALE 0.35355339059327379f   // 1/sqrt(8)
#define PBLK    128                      // partial blocks for BN reduction

typedef __attribute__((ext_vector_type(16))) _Float16 v16h;
typedef __attribute__((ext_vector_type(8)))  _Float16 v8h;
typedef __attribute__((ext_vector_type(4)))  _Float16 v4h;
typedef __attribute__((ext_vector_type(8)))  float    v8f;

// ---------------------------------------------------------------------------
// Atomics: positive-float max via int compare (all values > 0, init 0.0f),
// and hardware float add (global_atomic_add_f32 on gfx1250).
// ---------------------------------------------------------------------------
__device__ __forceinline__ void atomMaxPosF(float* p, float v) {
    atomicMax((int*)p, __float_as_int(v));
}

__device__ __forceinline__ void atomAddF(float* p, float v) {
    unsafeAtomicAdd(p, v);
}

// ---------------------------------------------------------------------------
// Zero-fill
// ---------------------------------------------------------------------------
__global__ void zero_kernel(float* __restrict__ p, int n) {
    int i = blockIdx.x * blockDim.x + threadIdx.x;
    if (i < n) p[i] = 0.0f;
}

// ---------------------------------------------------------------------------
// WMMA GEMM, fully compile-time shape + epilogue:
//   C[M x NCOLS] = A[M x Kdim] * B[Kdim x NCOLS]  (+bias, +resid, relu)
// Block: 128 threads (4 wave32). Block tile: 64 rows. Wave tile: 16 x NCOLS.
// A staged row-major f16 in LDS (sA[row][K], K zero-padded to KPAD);
// B staged TRANSPOSED f16 in LDS (sBt[col][K]) so every fragment is two
// contiguous 16-byte ds_load_b128s.
// Fragment layouts per CDNA5 ISA:
//   A 16x32 f16: lane m holds row m, K = {hi..hi+7, 16+hi..23+hi}, hi=(lane&16)?8:0
//   B 32x16 f16: lane n holds col n, K = {hi..hi+7, 16+hi..23+hi}, hi=(lane&16)?8:0
//   C f32 16x16: lane l: col=l&15, VGPR r -> row r + ((l&16)?8:0)
// ---------------------------------------------------------------------------
template<int KPAD, int NCOLS, bool BIAS, bool RESID, bool RELU>
__global__ __launch_bounds__(128)
void gemm_wmma_kernel(const float* __restrict__ A, const float* __restrict__ B,
                      const float* __restrict__ bias, const float* __restrict__ resid,
                      float* __restrict__ C, int M, int Kdim) {
    constexpr int NCT = NCOLS / 16;
    constexpr int KQ  = KPAD / 4;
    __shared__ _Float16 sA[64 * KPAD];
    __shared__ _Float16 sBt[NCOLS * KPAD];

    const int tid = threadIdx.x;
    const int rowBase = blockIdx.x * 64;

    // Stage A tile: float4 global loads -> 4-half LDS stores (Kdim % 4 == 0)
    for (int idx = tid; idx < 64 * KQ; idx += 128) {
        int r  = idx / KQ;
        int k4 = (idx - r * KQ) * 4;
        int gr = rowBase + r;
        if (gr >= M) gr = M - 1;               // clamp (stores are guarded)
        v4h o;
        if (k4 < Kdim) {
            const float4 v = *(const float4*)(A + (size_t)gr * Kdim + k4);
            o[0] = (_Float16)v.x; o[1] = (_Float16)v.y;
            o[2] = (_Float16)v.z; o[3] = (_Float16)v.w;
        } else {
            o[0] = (_Float16)0.0f; o[1] = (_Float16)0.0f;
            o[2] = (_Float16)0.0f; o[3] = (_Float16)0.0f;
        }
        *(v4h*)(sA + r * KPAD + k4) = o;
    }
    // Stage B transposed: sBt[n*KPAD + k] = B[k*NCOLS + n]
    for (int idx = tid; idx < NCOLS * KQ; idx += 128) {
        int n  = idx / KQ;
        int k4 = (idx - n * KQ) * 4;
        v4h o;
        if (k4 < Kdim) {
            o[0] = (_Float16)B[(size_t)(k4 + 0) * NCOLS + n];
            o[1] = (_Float16)B[(size_t)(k4 + 1) * NCOLS + n];
            o[2] = (_Float16)B[(size_t)(k4 + 2) * NCOLS + n];
            o[3] = (_Float16)B[(size_t)(k4 + 3) * NCOLS + n];
        } else {
            o[0] = (_Float16)0.0f; o[1] = (_Float16)0.0f;
            o[2] = (_Float16)0.0f; o[3] = (_Float16)0.0f;
        }
        *(v4h*)(sBt + n * KPAD + k4) = o;
    }
    __syncthreads();

    const int wave = tid >> 5;
    const int lane = tid & 31;

    v8f acc[NCT];
#pragma unroll
    for (int i = 0; i < NCT; ++i)
#pragma unroll
        for (int j = 0; j < 8; ++j)
            acc[i][j] = 0.0f;

    const int arow  = wave * 16 + (lane & 15);
    const int khi   = (lane & 16) ? 8 : 0;
    const int koffB = (lane & 16) ? 8 : 0;
    const int colL  = lane & 15;

#pragma unroll
    for (int kb = 0; kb < KPAD; kb += 32) {
        const _Float16* ap = sA + arow * KPAD + kb + khi;
        v8h a0 = *(const v8h*)(ap);
        v8h a1 = *(const v8h*)(ap + 16);
        v16h a = __builtin_shufflevector(a0, a1, 0, 1, 2, 3, 4, 5, 6, 7,
                                         8, 9, 10, 11, 12, 13, 14, 15);
#pragma unroll
        for (int ct = 0; ct < NCT; ++ct) {
            const _Float16* bp = sBt + (ct * 16 + colL) * KPAD + kb + koffB;
            v8h b0 = *(const v8h*)(bp);
            v8h b1 = *(const v8h*)(bp + 16);
            v16h b = __builtin_shufflevector(b0, b1, 0, 1, 2, 3, 4, 5, 6, 7,
                                             8, 9, 10, 11, 12, 13, 14, 15);
            acc[ct] = __builtin_amdgcn_wmma_f32_16x16x32_f16(
                false, a, false, b, (short)0, acc[ct], false, false);
        }
    }

    // Epilogue (compile-time fused bias / residual / relu)
    const int rOff = (lane & 16) ? 8 : 0;
    const int row0 = rowBase + wave * 16 + rOff;
    const bool fullBlock = (rowBase + 64 <= M);   // uniform across block

#pragma unroll
    for (int ct = 0; ct < NCT; ++ct) {
        const int col = ct * 16 + colL;
        float bv = 0.0f;
        if (BIAS) bv = bias[col];
        if (fullBlock) {
#pragma unroll
            for (int r = 0; r < 8; ++r) {
                size_t off = (size_t)(row0 + r) * NCOLS + col;
                float v = acc[ct][r] + bv;
                if (RESID) v += resid[off];
                if (RELU)  v = fmaxf(v, 0.0f);
                C[off] = v;
            }
        } else {
#pragma unroll
            for (int r = 0; r < 8; ++r) {
                if (row0 + r < M) {
                    size_t off = (size_t)(row0 + r) * NCOLS + col;
                    float v = acc[ct][r] + bv;
                    if (RESID) v += resid[off];
                    if (RELU)  v = fmaxf(v, 0.0f);
                    C[off] = v;
                }
            }
        }
    }
}

// ---------------------------------------------------------------------------
// Edge phase kernels. One thread per (edge, head).
// ---------------------------------------------------------------------------
// p = exp(clamp(dot(K[src,h], Q[dst,h]) / sqrt(dh), -10, 10));  m = segmax(p)
__global__ void sddmm_kernel(const float* __restrict__ Km, const float* __restrict__ Qm,
                             const int* __restrict__ src, const int* __restrict__ dst,
                             float* __restrict__ ebuf, float* __restrict__ mbuf) {
    int tid = blockIdx.x * blockDim.x + threadIdx.x;
    if (tid >= EE * HH) return;
    int ed = tid >> 3;
    int h  = tid & 7;
    int s = src[ed], d = dst[ed];
    const float* kp = Km + (size_t)s * DM + h * DHH;
    const float* qp = Qm + (size_t)d * DM + h * DHH;
    float acc = 0.0f;
#pragma unroll
    for (int i = 0; i < DHH; ++i) acc += kp[i] * qp[i];
    acc *= INV_SCALE;
    acc = fminf(fmaxf(acc, -10.0f), 10.0f);
    float p = expf(acc);
    ebuf[tid] = p;
    atomMaxPosF(&mbuf[d * HH + h], p);
}

// w = exp(p - m[dst]);  s = segsum(w)
__global__ void attn_norm_kernel(float* __restrict__ ebuf, const int* __restrict__ dst,
                                 const float* __restrict__ mbuf, float* __restrict__ sbuf) {
    int tid = blockIdx.x * blockDim.x + threadIdx.x;
    if (tid >= EE * HH) return;
    int ed = tid >> 3;
    int h  = tid & 7;
    int d = dst[ed];
    float w = expf(ebuf[tid] - mbuf[d * HH + h]);
    ebuf[tid] = w;
    atomAddF(&sbuf[d * HH + h], w);
}

// out[dst] += (w/s[dst]) * V[src]
__global__ void attn_agg_kernel(const float* __restrict__ ebuf, const int* __restrict__ src,
                                const int* __restrict__ dst, const float* __restrict__ sbuf,
                                const float* __restrict__ Vm, float* __restrict__ attn) {
    int tid = blockIdx.x * blockDim.x + threadIdx.x;
    if (tid >= EE * HH) return;
    int ed = tid >> 3;
    int h  = tid & 7;
    int s = src[ed], d = dst[ed];
    float alpha = ebuf[tid] / sbuf[d * HH + h];
    const float* vp = Vm + (size_t)s * DM + h * DHH;
    float* op = attn + (size_t)d * DM + h * DHH;
#pragma unroll
    for (int i = 0; i < DHH; ++i) atomAddF(&op[i], alpha * vp[i]);
}

// ---------------------------------------------------------------------------
// BatchNorm1d stats: deterministic two-stage, fully coalesced.
// Stage 1: 256-thread blocks; lane -> column (tid&63), row-quad -> tid>>6.
//   Each step reads 4 complete 64-float rows (coalesced), LDS-reduces the
//   4 row groups, writes per-block partials (no atomics).
// Stage 2: single block folds PBLK partials into stats[2*DM].
// ---------------------------------------------------------------------------
__global__ void bn_stats1_kernel(const float* __restrict__ x, float* __restrict__ part,
                                 int rows) {
    const int tid = threadIdx.x;
    const int c  = tid & (DM - 1);
    const int rq = tid >> 6;               // 0..3
    float s = 0.0f, s2 = 0.0f;
    for (int r = blockIdx.x * 4 + rq; r < rows; r += gridDim.x * 4) {
        float v = x[(size_t)r * DM + c];
        s += v; s2 += v * v;
    }
    __shared__ float sh[256], sh2[256];
    sh[tid] = s; sh2[tid] = s2;
    __syncthreads();
    if (tid < DM) {
        float ts  = sh[tid]  + sh[tid + 64]  + sh[tid + 128]  + sh[tid + 192];
        float ts2 = sh2[tid] + sh2[tid + 64] + sh2[tid + 128] + sh2[tid + 192];
        part[blockIdx.x * (2 * DM) + c]      = ts;
        part[blockIdx.x * (2 * DM) + DM + c] = ts2;
    }
}

__global__ void bn_stats2_kernel(const float* __restrict__ part, float* __restrict__ stats) {
    const int t = threadIdx.x;             // 0..127 -> stats[t]
    float acc = 0.0f;
    for (int b = 0; b < PBLK; ++b) acc += part[b * (2 * DM) + t];
    stats[t] = acc;
}

__global__ void bn_apply_kernel(const float* __restrict__ x, const float* __restrict__ stats,
                                const float* __restrict__ g, const float* __restrict__ b,
                                float* __restrict__ out, int rows) {
    int tid = blockIdx.x * blockDim.x + threadIdx.x;
    if (tid >= rows * DM) return;
    int c = tid & (DM - 1);
    const float invN = 1.0f / (float)rows;
    float mean = stats[c] * invN;
    float var  = stats[DM + c] * invN - mean * mean;
    out[tid] = (x[tid] - mean) * rsqrtf(var + EPSBN) * g[c] + b[c];
}

// ---------------------------------------------------------------------------
// Host orchestration
// ---------------------------------------------------------------------------
extern "C" void kernel_launch(void* const* d_in, const int* in_sizes, int n_in,
                              void* d_out, int out_size, void* d_ws, size_t ws_size,
                              hipStream_t stream) {
    const float* feat  = (const float*)d_in[0];
    const int*   src   = (const int*)  d_in[1];
    const int*   dst   = (const int*)  d_in[2];
    const float* W_emb = (const float*)d_in[3];
    const float* b_emb = (const float*)d_in[4];
    const float* Wq    = (const float*)d_in[5];
    const float* Wk    = (const float*)d_in[6];
    const float* Wv    = (const float*)d_in[7];
    const float* Wo    = (const float*)d_in[8];
    const float* bo    = (const float*)d_in[9];
    const float* bn1g  = (const float*)d_in[10];
    const float* bn1b  = (const float*)d_in[11];
    const float* W1    = (const float*)d_in[12];
    const float* b1    = (const float*)d_in[13];
    const float* W2    = (const float*)d_in[14];
    const float* b2    = (const float*)d_in[15];
    const float* bn2g  = (const float*)d_in[16];
    const float* bn2b  = (const float*)d_in[17];

    // Workspace layout (floats). attnb/mbuf/sbuf contiguous -> single zero.
    float* ws    = (float*)d_ws;
    float* hbuf  = ws;                        // N*D   (current h / residual)
    float* Qb    = hbuf  + (size_t)NN * DM;   // N*D
    float* Kb    = Qb    + (size_t)NN * DM;   // N*D
    float* Vb    = Kb    + (size_t)NN * DM;   // N*D
    float* attnb = Vb    + (size_t)NN * DM;   // N*D   (attn out / ffn2 out)
    float* mbuf  = attnb + (size_t)NN * DM;   // N*H
    float* sbuf  = mbuf  + (size_t)NN * HH;   // N*H
    float* tmp   = sbuf  + (size_t)NN * HH;   // max(N*2D, E*H): gemm tmp / ebuf
    size_t tmpSz = (size_t)NN * 2 * DM;
    if ((size_t)EE * HH > tmpSz) tmpSz = (size_t)EE * HH;
    float* ebuf  = tmp;                       // aliased (disjoint in time)
    float* stats = tmp   + tmpSz;             // 2*D
    float* part  = stats + 2 * DM;            // PBLK * 2*D

    const int gemmGrid = (NN + 63) / 64;
    const int edgeGrid = (EE * HH + 255) / 256;
    const int zeroN    = NN * DM + 2 * NN * HH;   // attnb + mbuf + sbuf

    // h = feat @ W_emb + b_emb   (K=16 padded to 32)
    gemm_wmma_kernel<32, 64, true, false, false><<<gemmGrid, 128, 0, stream>>>(
        feat, W_emb, b_emb, nullptr, hbuf, NN, D_IN);

    for (int l = 0; l < LL; ++l) {
        const float* Wq_l  = Wq  + (size_t)l * DM * DM;
        const float* Wk_l  = Wk  + (size_t)l * DM * DM;
        const float* Wv_l  = Wv  + (size_t)l * DM * DM;
        const float* Wo_l  = Wo  + (size_t)l * DM * DM;
        const float* bo_l  = bo  + (size_t)l * DM;
        const float* g1_l  = bn1g + (size_t)l * DM;
        const float* be1_l = bn1b + (size_t)l * DM;
        const float* W1_l  = W1  + (size_t)l * DM * 2 * DM;
        const float* b1_l  = b1  + (size_t)l * 2 * DM;
        const float* W2_l  = W2  + (size_t)l * 2 * DM * DM;
        const float* b2_l  = b2  + (size_t)l * DM;
        const float* g2_l  = bn2g + (size_t)l * DM;
        const float* be2_l = bn2b + (size_t)l * DM;

        // Projections (K=64, N=64)
        gemm_wmma_kernel<64, 64, false, false, false><<<gemmGrid, 128, 0, stream>>>(
            hbuf, Wq_l, nullptr, nullptr, Qb, NN, DM);
        gemm_wmma_kernel<64, 64, false, false, false><<<gemmGrid, 128, 0, stream>>>(
            hbuf, Wk_l, nullptr, nullptr, Kb, NN, DM);
        gemm_wmma_kernel<64, 64, false, false, false><<<gemmGrid, 128, 0, stream>>>(
            hbuf, Wv_l, nullptr, nullptr, Vb, NN, DM);

        // Zero segment accumulators (attnb | mbuf | sbuf, contiguous)
        zero_kernel<<<(zeroN + 255) / 256, 256, 0, stream>>>(attnb, zeroN);

        // Edge phase
        sddmm_kernel    <<<edgeGrid, 256, 0, stream>>>(Kb, Qb, src, dst, ebuf, mbuf);
        attn_norm_kernel<<<edgeGrid, 256, 0, stream>>>(ebuf, dst, mbuf, sbuf);
        attn_agg_kernel <<<edgeGrid, 256, 0, stream>>>(ebuf, src, dst, sbuf, Vb, attnb);

        // O-projection + residual, then BN1
        gemm_wmma_kernel<64, 64, true, true, false><<<gemmGrid, 128, 0, stream>>>(
            attnb, Wo_l, bo_l, hbuf, tmp, NN, DM);
        bn_stats1_kernel<<<PBLK, 256, 0, stream>>>(tmp, part, NN);
        bn_stats2_kernel<<<1, 128, 0, stream>>>(part, stats);
        bn_apply_kernel<<<((NN * DM) + 255) / 256, 256, 0, stream>>>(
            tmp, stats, g1_l, be1_l, hbuf, NN);

        // FFN: relu(h@W1+b1) @ W2 + b2 + residual, then BN2
        gemm_wmma_kernel<64, 128, true, false, true><<<gemmGrid, 128, 0, stream>>>(
            hbuf, W1_l, b1_l, nullptr, tmp, NN, DM);
        gemm_wmma_kernel<128, 64, true, true, false><<<gemmGrid, 128, 0, stream>>>(
            tmp, W2_l, b2_l, hbuf, attnb, NN, 2 * DM);
        bn_stats1_kernel<<<PBLK, 256, 0, stream>>>(attnb, part, NN);
        bn_stats2_kernel<<<1, 128, 0, stream>>>(part, stats);
        bn_apply_kernel<<<((NN * DM) + 255) / 256, 256, 0, stream>>>(
            attnb, stats, g2_l, be2_l, hbuf, NN);
    }

    hipMemcpyAsync(d_out, hbuf, (size_t)NN * DM * sizeof(float),
                   hipMemcpyDeviceToDevice, stream);
}
